// LinearTimeSelfAttention_39711267618933
// MI455X (gfx1250) — matrix operand order repaired and checked by reading, not compiled, not run on hardware
//
#include <hip/hip_runtime.h>

// ---------------- problem constants ----------------
#define NCH   256          // input channels (DIM)
#define NSP   65536        // spatial positions H*W
#define HID   128          // heads*dim_head
#define O3    384          // 3*HID
#define NHEAD 4
#define DHEAD 32
#define GSZ   (8 * 65536)  // elements per GroupNorm group (8 ch * 65536)
#define EPSV  1e-5f

// ---------------- workspace layout (bytes) ----------------
#define OFF_GSTAT 0u                      // 64 f32   : per-group {sum, sumsq}
#define OFF_KSTAT 1024u                   // 256 f32  : per-k-row {max, sumexp}
#define OFF_CTX   4096u                   // 4096 f32 : context [head][d][e]
#define OFF_WQKV  32768u                  // 98304 f16: qkv weights
#define OFF_W2    262144u                 // 32768 f16: folded out weights
#define OFF_QH    524288u                 // 128*65536 f16 : q
#define OFF_KH    (OFF_QH + 16777216u)    // k
#define OFF_VH    (OFF_KH + 16777216u)    // v

typedef __attribute__((ext_vector_type(16))) _Float16 v16h;
typedef __attribute__((ext_vector_type(8)))  float    v8f;

// ---- WMMA fragment loaders (CDNA5 16-bit layouts, wave32) ----
// A matrix 16x32 f16, source row-major [m][k] with leading dim ld.
// lane: m = m0 + (lane&15); K base += 8 if lane>=16.
// elements 0..7 -> K = kb..kb+7 (VGPR0-3), 8..15 -> K = kb+16..kb+23 (VGPR4-7)
__device__ inline v16h load_fragA(const _Float16* __restrict__ A, int m0, int k0,
                                  int ld, int lane) {
  const _Float16* p = A + (size_t)(m0 + (lane & 15)) * (size_t)ld + k0 + ((lane & 16) ? 8 : 0);
  v16h a;
#pragma unroll
  for (int i = 0; i < 8; ++i) a[i] = p[i];
#pragma unroll
  for (int i = 0; i < 8; ++i) a[8 + i] = p[16 + i];
  return a;
}

// B matrix 32x16 f16, source stored "per-column K-contiguous": Bsrc[n][k] with ld.
// lane: n = n0 + (lane&15); K base += 8 if lane>=16; elements 0..7 -> K = kb..kb+7, 8..15 -> K = kb+16..kb+23
__device__ inline v16h load_fragB(const _Float16* __restrict__ Bsrc, int n0, int k0,
                                  int ld, int lane) {
  const _Float16* p = Bsrc + (size_t)(n0 + (lane & 15)) * (size_t)ld + k0 + ((lane & 16) ? 8 : 0);
  v16h b;
#pragma unroll
  for (int i = 0; i < 8; ++i) b[i] = p[i];
#pragma unroll
  for (int i = 0; i < 8; ++i) b[8 + i] = p[16 + i];
  return b;
}

// ---------------- kernel 0: init workspace + convert qkv weights to f16 ----------------
__global__ __launch_bounds__(256) void k_init(const float* __restrict__ wq_f32,
                                              _Float16* __restrict__ wqh,
                                              float* __restrict__ gstat,
                                              float* __restrict__ ctx) {
  int i = blockIdx.x * 256 + threadIdx.x;
  if (i < O3 * NCH) wqh[i] = (_Float16)wq_f32[i];
  if (i < 64) gstat[i] = 0.0f;
  if (i < 4096) ctx[i] = 0.0f;
}

// ---------------- kernel 1: GroupNorm statistics (sum, sumsq per group) ----------------
__global__ __launch_bounds__(256) void k_gstat(const float* __restrict__ x,
                                               float* __restrict__ gstat) {
  __shared__ float rs[256];
  __shared__ float rq[256];
  int g = blockIdx.y;
  size_t base = (size_t)g * GSZ + (size_t)blockIdx.x * (GSZ / 32);
  float s = 0.0f, q = 0.0f;
  for (int i = threadIdx.x; i < GSZ / 32; i += 256) {
    float v = x[base + i];
    s += v;
    q += v * v;
  }
  rs[threadIdx.x] = s;
  rq[threadIdx.x] = q;
  __syncthreads();
  for (int st = 128; st > 0; st >>= 1) {
    if ((int)threadIdx.x < st) {
      rs[threadIdx.x] += rs[threadIdx.x + st];
      rq[threadIdx.x] += rq[threadIdx.x + st];
    }
    __syncthreads();
  }
  if (threadIdx.x == 0) {
    atomicAdd(&gstat[2 * g], rs[0]);
    atomicAdd(&gstat[2 * g + 1], rq[0]);
  }
}

// ---------------- kernel 2: fused GroupNorm + QKV GEMM (WMMA) ----------------
// out tile: 64 rows (of 384) x 64 cols; x tile normalized & transposed into LDS [n][c]
__global__ __launch_bounds__(256) void k_qkv(const float* __restrict__ x,
                                             const float* __restrict__ gnw,
                                             const float* __restrict__ gnb,
                                             const float* __restrict__ qkvb,
                                             const float* __restrict__ gstat,
                                             const _Float16* __restrict__ wq,
                                             _Float16* __restrict__ qh,
                                             _Float16* __restrict__ kh,
                                             _Float16* __restrict__ vh) {
  __shared__ _Float16 xs[64][NCH + 8];
  __shared__ float ssh[NCH];
  __shared__ float tsh[NCH];
  int tid = threadIdx.x;
  {  // per-channel affine fold of GroupNorm
    int c = tid;
    int g = c >> 3;
    float mean = gstat[2 * g] * (1.0f / (float)GSZ);
    float var = gstat[2 * g + 1] * (1.0f / (float)GSZ) - mean * mean;
    float rinv = rsqrtf(var + EPSV);
    float s = rinv * gnw[c];
    ssh[c] = s;
    tsh[c] = gnb[c] - mean * s;
  }
  __syncthreads();
  int nb = blockIdx.x * 64;
  int n = tid & 63;
  int c0 = tid >> 6;
  for (int c = c0; c < NCH; c += 4) {
    float v = x[(size_t)c * NSP + nb + n];
    xs[n][c] = (_Float16)(v * ssh[c] + tsh[c]);
  }
  __syncthreads();

  int wave = tid >> 5, lane = tid & 31;
  int mt = blockIdx.y * 64 + (wave & 3) * 16;
  int wn2 = wave >> 2;
#pragma unroll
  for (int j = 0; j < 2; ++j) {
    int ncol = (wn2 + 2 * j) * 16;
    v8f acc = {};
#pragma unroll
    for (int k = 0; k < NCH; k += 32) {
      v16h a = load_fragA(wq, mt, k, NCH, lane);
      v16h b = load_fragB(&xs[0][0], ncol, k, NCH + 8, lane);
      acc = __builtin_amdgcn_wmma_f32_16x16x32_f16(false, a, false, b, (short)0, acc,
                                                   false, false);
    }
    int nidx = nb + ncol + (lane & 15);
    int mhi = (lane & 16) ? 8 : 0;
#pragma unroll
    for (int r = 0; r < 8; ++r) {
      int row = mt + r + mhi;
      float v = acc[r] + qkvb[row];
      if (row < HID)
        qh[(size_t)row * NSP + nidx] = (_Float16)v;
      else if (row < 2 * HID)
        kh[(size_t)(row - HID) * NSP + nidx] = (_Float16)v;
      else
        vh[(size_t)(row - 2 * HID) * NSP + nidx] = (_Float16)v;
    }
  }
}

// ---------------- kernel 3: per-row softmax stats of k (max, sum exp) ----------------
__global__ __launch_bounds__(256) void k_ksm(const _Float16* __restrict__ kh,
                                             float* __restrict__ kstat) {
  __shared__ float red[256];
  int row = blockIdx.x, tid = threadIdx.x;
  const _Float16* p = kh + (size_t)row * NSP;
  float m = -1e30f;
  for (int i = tid; i < NSP; i += 256) m = fmaxf(m, (float)p[i]);
  red[tid] = m;
  __syncthreads();
  for (int s = 128; s > 0; s >>= 1) {
    if (tid < s) red[tid] = fmaxf(red[tid], red[tid + s]);
    __syncthreads();
  }
  m = red[0];
  __syncthreads();
  float sum = 0.0f;
  for (int i = tid; i < NSP; i += 256) sum += __expf((float)p[i] - m);
  red[tid] = sum;
  __syncthreads();
  for (int s = 128; s > 0; s >>= 1) {
    if (tid < s) red[tid] += red[tid + s];
    __syncthreads();
  }
  if (tid == 0) {
    kstat[2 * row] = m;
    kstat[2 * row + 1] = red[0];
  }
}

// ---------------- kernel 4: context = exp(k-max) @ v^T  (WMMA reduction over n) ----------------
// grid: (8 n-chunks, 4 heads); each wave handles a 1024-wide n-slice, 2x2 16x16 tiles
__global__ __launch_bounds__(256) void k_ctx(const _Float16* __restrict__ kh,
                                             const _Float16* __restrict__ vh,
                                             const float* __restrict__ kstat,
                                             float* __restrict__ ctx) {
  __shared__ float cbuf[1024];
  for (int i = threadIdx.x; i < 1024; i += 256) cbuf[i] = 0.0f;
  __syncthreads();

  int head = blockIdx.y;
  int wave = threadIdx.x >> 5, lane = threadIdx.x & 31;
  int nbase = blockIdx.x * 8192 + wave * 1024;

  v8f acc[2][2] = {{{}, {}}, {{}, {}}};
  for (int nk = nbase; nk < nbase + 1024; nk += 32) {
    v16h afrag[2], bfrag[2];
#pragma unroll
    for (int dt = 0; dt < 2; ++dt) {
      int row = head * DHEAD + dt * 16 + (lane & 15);
      float rmax = kstat[2 * row];
      v16h a = load_fragA(kh, head * DHEAD + dt * 16, nk, NSP, lane);
#pragma unroll
      for (int i = 0; i < 16; ++i) a[i] = (_Float16)__expf((float)a[i] - rmax);
      afrag[dt] = a;
    }
#pragma unroll
    for (int et = 0; et < 2; ++et)
      bfrag[et] = load_fragB(vh, head * DHEAD + et * 16, nk, NSP, lane);
#pragma unroll
    for (int dt = 0; dt < 2; ++dt)
#pragma unroll
      for (int et = 0; et < 2; ++et)
        acc[dt][et] = __builtin_amdgcn_wmma_f32_16x16x32_f16(
            false, afrag[dt], false, bfrag[et], (short)0, acc[dt][et], false, false);
  }

  int mhi = (lane & 16) ? 8 : 0;
  int elo = lane & 15;
#pragma unroll
  for (int dt = 0; dt < 2; ++dt)
#pragma unroll
    for (int et = 0; et < 2; ++et)
#pragma unroll
      for (int r = 0; r < 8; ++r) {
        int d = dt * 16 + r + mhi;
        int e = et * 16 + elo;
        atomicAdd(&cbuf[d * 32 + e], acc[dt][et][r]);  // LDS atomic
      }
  __syncthreads();
  for (int i = threadIdx.x; i < 1024; i += 256)
    atomicAdd(&ctx[head * 1024 + i], cbuf[i]);
}

// ---------------- kernel 5: fold context + softmax norm into out_weight -> W2 (f16) ----------------
__global__ __launch_bounds__(256) void k_fold(const float* __restrict__ ow,
                                              const float* __restrict__ ctx,
                                              const float* __restrict__ kstat,
                                              _Float16* __restrict__ w2) {
  int o = threadIdx.x;  // 0..255
  for (int hd = 0; hd < HID; ++hd) {
    int h = hd >> 5, d = hd & 31;
    float acc = 0.0f;
#pragma unroll
    for (int e = 0; e < DHEAD; ++e)
      acc += ow[o * HID + h * DHEAD + e] * ctx[h * 1024 + d * 32 + e];
    w2[o * HID + hd] = (_Float16)(acc / kstat[2 * hd + 1]);
  }
}

// ---------------- kernel 6: out = W2 @ q + bias (WMMA), f32 output ----------------
__global__ __launch_bounds__(256) void k_out(const _Float16* __restrict__ qh,
                                             const _Float16* __restrict__ w2,
                                             const float* __restrict__ ob,
                                             float* __restrict__ out) {
  __shared__ _Float16 qs[64][HID + 8];
  int tid = threadIdx.x;
  int nb = blockIdx.x * 64;
  int n = tid & 63, c0 = tid >> 6;
  for (int c = c0; c < HID; c += 4) qs[n][c] = qh[(size_t)c * NSP + nb + n];
  __syncthreads();

  int wave = tid >> 5, lane = tid & 31;
  int mt = blockIdx.y * 128 + wave * 16;
#pragma unroll
  for (int j = 0; j < 4; ++j) {
    int ncol = j * 16;
    v8f acc = {};
#pragma unroll
    for (int k = 0; k < HID; k += 32) {
      v16h a = load_fragA(w2, mt, k, HID, lane);
      v16h b = load_fragB(&qs[0][0], ncol, k, HID + 8, lane);
      acc = __builtin_amdgcn_wmma_f32_16x16x32_f16(false, a, false, b, (short)0, acc,
                                                   false, false);
    }
    int nidx = nb + ncol + (lane & 15);
    int mhi = (lane & 16) ? 8 : 0;
#pragma unroll
    for (int r = 0; r < 8; ++r) {
      int row = mt + r + mhi;
      out[(size_t)row * NSP + nidx] = acc[r] + ob[row];
    }
  }
}

// ---------------- host launch ----------------
extern "C" void kernel_launch(void* const* d_in, const int* in_sizes, int n_in,
                              void* d_out, int out_size, void* d_ws, size_t ws_size,
                              hipStream_t stream) {
  const float* x = (const float*)d_in[0];
  const float* gn_weight = (const float*)d_in[1];
  const float* gn_bias = (const float*)d_in[2];
  const float* qkv_weight = (const float*)d_in[3];
  const float* qkv_bias = (const float*)d_in[4];
  const float* out_weight = (const float*)d_in[5];
  const float* out_bias = (const float*)d_in[6];
  float* out = (float*)d_out;

  char* ws = (char*)d_ws;
  float* gstat = (float*)(ws + OFF_GSTAT);
  float* kstat = (float*)(ws + OFF_KSTAT);
  float* ctx = (float*)(ws + OFF_CTX);
  _Float16* wqh = (_Float16*)(ws + OFF_WQKV);
  _Float16* w2 = (_Float16*)(ws + OFF_W2);
  _Float16* qh = (_Float16*)(ws + OFF_QH);
  _Float16* kh = (_Float16*)(ws + OFF_KH);
  _Float16* vh = (_Float16*)(ws + OFF_VH);

  // 0) init workspace + convert qkv weights to f16
  k_init<<<dim3((O3 * NCH + 255) / 256), dim3(256), 0, stream>>>(qkv_weight, wqh, gstat, ctx);
  // 1) GroupNorm stats
  k_gstat<<<dim3(32, 32), dim3(256), 0, stream>>>(x, gstat);
  // 2) fused GN + QKV GEMM -> q,k,v (f16)
  k_qkv<<<dim3(NSP / 64, O3 / 64), dim3(256), 0, stream>>>(x, gn_weight, gn_bias, qkv_bias,
                                                           gstat, wqh, qh, kh, vh);
  // 3) k softmax row stats
  k_ksm<<<dim3(HID), dim3(256), 0, stream>>>(kh, kstat);
  // 4) context reduction (WMMA)
  k_ctx<<<dim3(NSP / 8192, NHEAD), dim3(256), 0, stream>>>(kh, vh, kstat, ctx);
  // 5) fold context into out projection -> W2
  k_fold<<<dim3(1), dim3(256), 0, stream>>>(out_weight, ctx, kstat, w2);
  // 6) out = W2 @ q + bias
  k_out<<<dim3(NSP / 64, 2), dim3(256), 0, stream>>>(qh, w2, out_bias, out);
}
